// LSTM_26147760898060
// MI455X (gfx1250) — matrix operand order repaired and checked by reading, not compiled, not run on hardware
//
#include <hip/hip_runtime.h>
#include <hip/hip_bf16.h>
#include <cstddef>
#include <cstdint>

// LSTM: T=512, B=128, I=256, H=256.  wave32 / WGP / WMMA (gfx1250).
#define T_STEPS 512
#define B_DIM   128
#define I_DIM   256
#define H_DIM   256

typedef __attribute__((ext_vector_type(16))) __bf16 v16bf;
typedef __attribute__((ext_vector_type(8)))  __bf16 v8bf;
typedef __attribute__((ext_vector_type(8)))  float  v8f;

// ---------------------------------------------------------------------------
// WMMA operand layouts used here (wave32, contiguous k order for A and B), mapped to row-major memory:
//
// A (16x32 bf16, M=rows, K=contraction), per lane = 16 bf16 in 8 VGPRs:
//   lane<16 : M=lane,    K = k0*32 + 0..15
//   lane>=16: M=lane-16, K = k0*32 + 16..31
//   -> one contiguous 16-element (32B) run (two 16B loads at kbase, kbase+8),
//      kbase = k0*32 + (lane>>4)*16, from a row-major [row][K] matrix.
//
// B (32x16 bf16, B[k][n] = W[n][k]), per lane = 16 bf16:
//   lane<16 : N=lane,    K = k0*32 + 0..15
//   lane>=16: N=lane-16, K = k0*32 + 16..31
//   -> one contiguous 16-element (32B) run of row N of row-major W.
//
// C/D f32 16x16: lane<16: N=lane, M=0..7 over 8 regs; lane>=16: N=lane-16, M=8..15.
// ---------------------------------------------------------------------------

__device__ __forceinline__ v16bf load_a16(const __bf16* p) {
    v8bf lo = *(const v8bf*)(p);        // K 0..7 of half (16B)
    v8bf hi = *(const v8bf*)(p + 8);    // K 8..15 of half (16B)
    return __builtin_shufflevector(lo, hi, 0, 1, 2, 3, 4, 5, 6, 7,
                                           8, 9, 10, 11, 12, 13, 14, 15);
}

__device__ __forceinline__ void async_copy_b128(unsigned lds_off, const void* gaddr) {
    // GLOBAL_LOAD_ASYNC_TO_LDS_B128 (GV mode): VDST = LDS byte address VGPR,
    // VADDR = 64-bit global address. Tracked by ASYNCcnt.
    asm volatile("global_load_async_to_lds_b128 %0, %1, off"
                 :: "v"(lds_off), "v"(gaddr)
                 : "memory");
}

__device__ __forceinline__ void wait_async0() {
    asm volatile("s_wait_asynccnt 0x0" ::: "memory");
}

__device__ __forceinline__ float sigmoidf_(float x) {
    return 1.0f / (1.0f + __expf(-x));
}

// ---------------------------------------------------------------------------
// One-time conversions (all weights are tiny and L2-resident afterwards).
// ---------------------------------------------------------------------------
__global__ __launch_bounds__(256) void convert_x_kernel(
    const float* __restrict__ x, __bf16* __restrict__ xbf)
{
    size_t tid = (size_t)blockIdx.x * blockDim.x + threadIdx.x; // 16,777,216
    xbf[tid] = (__bf16)x[tid];
}

__global__ __launch_bounds__(256) void convert_w_kernel(
    const float* __restrict__ Wi, const float* __restrict__ Wf,
    const float* __restrict__ Wc, const float* __restrict__ Wo,
    const float* __restrict__ Ui, const float* __restrict__ Uf,
    const float* __restrict__ Uc, const float* __restrict__ Uo,
    __bf16* __restrict__ dst)
{
    unsigned tid = blockIdx.x * blockDim.x + threadIdx.x;       // 8 * 65536
    unsigned m = tid >> 16;
    unsigned r = tid & 65535u;
    const float* src;
    switch (m) {
        case 0: src = Wi; break; case 1: src = Wf; break;
        case 2: src = Wc; break; case 3: src = Wo; break;
        case 4: src = Ui; break; case 5: src = Uf; break;
        case 6: src = Uc; break; default: src = Uo; break;
    }
    dst[tid] = (__bf16)src[r];
}

__global__ __launch_bounds__(256) void bias_kernel(
    const float* __restrict__ bih, const float* __restrict__ bhh,
    float* __restrict__ bias)
{
    unsigned tid = blockIdx.x * blockDim.x + threadIdx.x;       // 1024
    bias[tid] = bih[tid] + bhh[tid];
}

// ---------------------------------------------------------------------------
// Persistent LSTM scan. Grid = 8 blocks (one 16-row batch tile each),
// 512 threads = 16 waves; wave n0 owns the 16x16 (batch x hidden) tile
// (m0 = blockIdx, n0). Batch tiles are fully independent across the whole
// sequence, so the carried state never leaves the block:
//   h (bf16, WMMA A source)  -> LDS, republished each step
//   h,c (fp32, exact carry)  -> VGPRs (v8f per wave)
// Per step per wave: 64 v_wmma_f32_16x16x32_bf16 (x path + h path, 4 gates),
// while the next step's x slice is async-DMA'd into the alternate LDS buffer.
// ---------------------------------------------------------------------------
__global__ __launch_bounds__(512) void lstm_scan_kernel(
    const __bf16* __restrict__ xbf,   // [T][B][I] bf16 row-major
    const __bf16* __restrict__ wbf,   // W_i,W_f,W_c,W_o each [H][I] bf16
    const __bf16* __restrict__ ubf,   // U_i,U_f,U_c,U_o each [H][H] bf16
    const float*  __restrict__ bias,  // [4*H] summed bias
    const int*    __restrict__ lengths,
    float* __restrict__ out,          // [T][B][H]
    float* __restrict__ hf,           // [B][H]
    float* __restrict__ cf)           // [B][H]
{
    __shared__ __bf16 hsh[16 * 256];      // carried h slice (8KB)
    __shared__ __bf16 xsh[2][16 * 256];   // double-buffered x slice (16KB)

    const int tid  = threadIdx.x;
    const int lane = tid & 31;
    const int n0   = tid >> 5;        // wave id = hidden tile 0..15
    const int m0   = blockIdx.x;      // batch tile 0..7

    // --- prologue: h = 0 in LDS; stage x[0] via async DMA ------------------
    #pragma unroll
    for (int i = 0; i < 8; ++i) hsh[tid * 8 + i] = (__bf16)0.0f;
    {
        unsigned lds_off = (unsigned)(size_t)&xsh[0][tid * 8];
        const void* g = xbf + ((size_t)0 * B_DIM + m0 * 16) * I_DIM + tid * 8;
        async_copy_b128(lds_off, g);
    }
    wait_async0();
    __syncthreads();

    // --- per-lane constants -------------------------------------------------
    const int col     = lane & 15;
    const int rowbase = (lane >> 4) * 8;      // C/D: M rows 0..7 or 8..15
    const int arow    = lane & 15;            // A: M row within tile
    const int asel    = (lane >> 4) * 16;     // A: K half select
    const int j       = n0 * 16 + col;        // hidden index
    const float bi = bias[0 * H_DIM + j];
    const float bf = bias[1 * H_DIM + j];
    const float bc = bias[2 * H_DIM + j];
    const float bo = bias[3 * H_DIM + j];
    int len[8];
    #pragma unroll
    for (int r = 0; r < 8; ++r) len[r] = lengths[m0 * 16 + rowbase + r];

    // B-operand bases: row N = j of each row-major weight matrix.
    const __bf16* wrow = wbf + (size_t)j * I_DIM + (lane >> 4) * 16;
    const __bf16* urow = ubf + (size_t)j * H_DIM + (lane >> 4) * 16;

    v8f h_reg = {};   // carried h (fp32, exact)
    v8f c_reg = {};   // carried c (fp32, exact)

    for (int t = 0; t < T_STEPS; ++t) {
        const int cur = t & 1;

        // Prefetch next x slice into the alternate LDS buffer (ASYNCcnt).
        if (t + 1 < T_STEPS) {
            unsigned lds_off = (unsigned)(size_t)&xsh[cur ^ 1][tid * 8];
            const void* g = xbf + ((size_t)(t + 1) * B_DIM + m0 * 16) * I_DIM + tid * 8;
            async_copy_b128(lds_off, g);
        }

        v8f acc_i = {}, acc_f = {}, acc_c = {}, acc_o = {};
        const __bf16* xrow = &xsh[cur][arow * 256];
        const __bf16* hrow = &hsh[arow * 256];

        #pragma unroll
        for (int k0 = 0; k0 < 8; ++k0) {
            const int abase = k0 * 32 + asel;
            // x[t] @ W_g^T   (A from LDS, B from global/L2)
            v16bf ax = load_a16(xrow + abase);
            v16bf bw0 = *(const v16bf*)(wrow + 0 * 65536 + k0 * 32);
            v16bf bw1 = *(const v16bf*)(wrow + 1 * 65536 + k0 * 32);
            v16bf bw2 = *(const v16bf*)(wrow + 2 * 65536 + k0 * 32);
            v16bf bw3 = *(const v16bf*)(wrow + 3 * 65536 + k0 * 32);
            acc_i = __builtin_amdgcn_wmma_f32_16x16x32_bf16(false, ax, false, bw0, (short)0, acc_i, false, false);
            acc_f = __builtin_amdgcn_wmma_f32_16x16x32_bf16(false, ax, false, bw1, (short)0, acc_f, false, false);
            acc_c = __builtin_amdgcn_wmma_f32_16x16x32_bf16(false, ax, false, bw2, (short)0, acc_c, false, false);
            acc_o = __builtin_amdgcn_wmma_f32_16x16x32_bf16(false, ax, false, bw3, (short)0, acc_o, false, false);
            // h_{t-1} @ U_g^T
            v16bf ah = load_a16(hrow + abase);
            v16bf bu0 = *(const v16bf*)(urow + 0 * 65536 + k0 * 32);
            v16bf bu1 = *(const v16bf*)(urow + 1 * 65536 + k0 * 32);
            v16bf bu2 = *(const v16bf*)(urow + 2 * 65536 + k0 * 32);
            v16bf bu3 = *(const v16bf*)(urow + 3 * 65536 + k0 * 32);
            acc_i = __builtin_amdgcn_wmma_f32_16x16x32_bf16(false, ah, false, bu0, (short)0, acc_i, false, false);
            acc_f = __builtin_amdgcn_wmma_f32_16x16x32_bf16(false, ah, false, bu1, (short)0, acc_f, false, false);
            acc_c = __builtin_amdgcn_wmma_f32_16x16x32_bf16(false, ah, false, bu2, (short)0, acc_c, false, false);
            acc_o = __builtin_amdgcn_wmma_f32_16x16x32_bf16(false, ah, false, bu3, (short)0, acc_o, false, false);
        }

        // --- fused gate math + masked state update (fp32) -------------------
        float hn_new[8];
        float* out_t = out + (size_t)t * B_DIM * H_DIM;
        #pragma unroll
        for (int r = 0; r < 8; ++r) {
            const int b   = m0 * 16 + rowbase + r;
            const int idx = b * H_DIM + j;
            const float msk = (t < len[r]) ? 1.0f : 0.0f;
            const float iv = sigmoidf_(acc_i[r] + bi);
            const float fv = sigmoidf_(acc_f[r] + bf);
            const float gv = tanhf    (acc_c[r] + bc);
            const float ov = sigmoidf_(acc_o[r] + bo);
            const float cn = fv * c_reg[r] + iv * gv;
            const float hn = ov * tanhf(cn);
            h_reg[r] = msk * hn + (1.0f - msk) * h_reg[r];
            c_reg[r] = msk * cn + (1.0f - msk) * c_reg[r];
            hn_new[r] = h_reg[r];
            out_t[idx] = hn * msk;
        }

        __syncthreads();   // everyone finished reading hsh / xsh[cur]
        #pragma unroll
        for (int r = 0; r < 8; ++r)
            hsh[(rowbase + r) * 256 + j] = (__bf16)hn_new[r];
        wait_async0();     // our prefetch into xsh[cur^1] landed
        __syncthreads();   // new h + new x visible to all waves
    }

    // --- epilogue: final h, c ----------------------------------------------
    #pragma unroll
    for (int r = 0; r < 8; ++r) {
        const int idx = (m0 * 16 + rowbase + r) * H_DIM + j;
        hf[idx] = h_reg[r];
        cf[idx] = c_reg[r];
    }
}

extern "C" void kernel_launch(void* const* d_in, const int* in_sizes, int n_in,
                              void* d_out, int out_size, void* d_ws, size_t ws_size,
                              hipStream_t stream) {
    (void)in_sizes; (void)n_in; (void)out_size; (void)ws_size;

    const float* x       = (const float*)d_in[0];
    const int*   lengths = (const int*)  d_in[1];
    const float* W_i     = (const float*)d_in[2];
    const float* U_i     = (const float*)d_in[3];
    const float* W_f     = (const float*)d_in[4];
    const float* U_f     = (const float*)d_in[5];
    const float* W_c     = (const float*)d_in[6];
    const float* U_c     = (const float*)d_in[7];
    const float* W_o     = (const float*)d_in[8];
    const float* U_o     = (const float*)d_in[9];
    const float* bias_ih = (const float*)d_in[10];
    const float* bias_hh = (const float*)d_in[11];

    float* out    = (float*)d_out;
    float* out_hf = out + (size_t)T_STEPS * B_DIM * H_DIM;
    float* out_cf = out_hf + (size_t)B_DIM * H_DIM;

    // Workspace: xbf (32MB) | wbf+ubf (1MB) | bias (4KB)
    char*   ws    = (char*)d_ws;
    __bf16* xbf   = (__bf16*)ws;
    __bf16* wbf   = (__bf16*)(ws + (size_t)T_STEPS * B_DIM * I_DIM * 2);
    __bf16* ubf   = wbf + 4 * H_DIM * I_DIM;
    float*  biasp = (float*)((char*)(ubf + 4 * H_DIM * H_DIM));

    convert_x_kernel<<<65536, 256, 0, stream>>>(x, xbf);
    convert_w_kernel<<<2048, 256, 0, stream>>>(W_i, W_f, W_c, W_o,
                                               U_i, U_f, U_c, U_o, wbf);
    bias_kernel<<<4, 256, 0, stream>>>(bias_ih, bias_hh, biasp);

    lstm_scan_kernel<<<8, 512, 0, stream>>>(xbf, wbf, ubf, biasp, lengths,
                                            out, out_hf, out_cf);
}
